// RWKV6_90572270338517
// MI455X (gfx1250) — compile-verified, matrix operand order checked by reading, not run on hardware
//
#include <hip/hip_runtime.h>
#include <hip/hip_bf16.h>

// ---------------------------------------------------------------------------
// RWKV6 forward for MI455X (gfx1250, wave32, WMMA)
// B=4 T=512 C=2048 H=32 N=64 ; D_MIX=32 D_DEC=64
// ---------------------------------------------------------------------------

typedef __attribute__((ext_vector_type(16))) __bf16 v16bf;
typedef __attribute__((ext_vector_type(8)))  float  v8f;
typedef int v4i_vs __attribute__((vector_size(16)));   // matches async-LDS builtin param

union FragU { uint4 u[2]; v16bf v; };

static constexpr int Bc = 4;
static constexpr int Tc = 512;
static constexpr int Cc = 2048;
static constexpr int Hc = 32;
static constexpr int Nh = 64;
static constexpr int Mrows = Bc * Tc;                 // 2048
static constexpr int OUT_ELEMS   = Bc * Tc * Cc;      // 4,194,304
static constexpr int SHIFT_ELEMS = Bc * Cc;           // 8,192

// ---- optional gfx1250 async Global->LDS staging (compile-safe guards) ----
#if defined(__has_builtin)
#  if __has_builtin(__builtin_amdgcn_global_load_async_to_lds_b128)
#    define HAS_ASYNC_LDS 1
#  endif
#endif
#ifndef HAS_ASYNC_LDS
#  define HAS_ASYNC_LDS 0
#endif

__device__ __forceinline__ void wait_asynccnt0() {
#if defined(__has_builtin) && __has_builtin(__builtin_amdgcn_s_wait_asynccnt)
    __builtin_amdgcn_s_wait_asynccnt(0);
#else
    asm volatile("s_wait_asynccnt 0x0" ::: "memory");
#endif
}

// ============================================================
// Kernel 0: transpose + convert weights: Wt[n][k] = bf16(W[k][n])
// 32x32 LDS tiles, grid.z selects one of the 5 weights
// ============================================================
__global__ __launch_bounds__(256)
void wtrans_kernel(const float* __restrict__ W0, const float* __restrict__ W1,
                   const float* __restrict__ W2, const float* __restrict__ W3,
                   const float* __restrict__ W4,
                   __hip_bfloat16* __restrict__ Wt_all)
{
    const float* Ws[5] = { W0, W1, W2, W3, W4 };
    const float* W = Ws[blockIdx.z];
    __hip_bfloat16* Wt = Wt_all + (size_t)blockIdx.z * Cc * Cc;

    __shared__ float tile[32][33];
    const int k0 = blockIdx.y * 32;
    const int n0 = blockIdx.x * 32;
    const int tx = threadIdx.x & 31;
    const int ty = threadIdx.x >> 5;      // 0..7

    #pragma unroll
    for (int i = 0; i < 32; i += 8)
        tile[ty + i][tx] = W[(size_t)(k0 + ty + i) * Cc + n0 + tx];
    __syncthreads();
    #pragma unroll
    for (int i = 0; i < 32; i += 8)
        Wt[(size_t)(n0 + ty + i) * Cc + k0 + tx] = __float2bfloat16(tile[tx][ty + i]);
}

// ============================================================
// Kernel 1: token shift + maa mixing -> 5 bf16 activation bufs
// ============================================================
__global__ __launch_bounds__(256)
void rwkv_mix_kernel(const float* __restrict__ x,
                     const float* __restrict__ shift_state,
                     const float* __restrict__ maa_x,
                     const float* __restrict__ maa_w,
                     const float* __restrict__ maa_k,
                     const float* __restrict__ maa_v,
                     const float* __restrict__ maa_r,
                     const float* __restrict__ maa_g,
                     const float* __restrict__ maa_w1,   // (C, 160)
                     const float* __restrict__ maa_w2,   // (5, 32, C)
                     __hip_bfloat16* __restrict__ xw,
                     __hip_bfloat16* __restrict__ xk,
                     __hip_bfloat16* __restrict__ xv,
                     __hip_bfloat16* __restrict__ xr,
                     __hip_bfloat16* __restrict__ xg)
{
    __shared__ float xs[Cc];
    __shared__ float xxs[Cc];
    __shared__ float zs[Cc];
    __shared__ float xxx[160];

    const int row = blockIdx.x;            // b*T + t
    const int b   = row / Tc;
    const int t   = row % Tc;
    const int tid = threadIdx.x;

    const float* xrow = x + (size_t)row * Cc;
    const float* prow = (t == 0) ? (shift_state + (size_t)b * Cc)
                                 : (x + (size_t)(row - 1) * Cc);

    for (int c = tid; c < Cc; c += 256) {
        float xv_ = xrow[c];
        float xx  = prow[c] - xv_;
        xs[c]  = xv_;
        xxs[c] = xx;
        zs[c]  = xv_ + xx * maa_x[c];
    }
    __syncthreads();

    if (tid < 160) {
        float s = 0.f;
        for (int c = 0; c < Cc; ++c)
            s += zs[c] * maa_w1[(size_t)c * 160 + tid];
        xxx[tid] = tanhf(s);
    }
    __syncthreads();

    const float* maas[5] = { maa_w, maa_k, maa_v, maa_r, maa_g };
    __hip_bfloat16* outs[5] = { xw, xk, xv, xr, xg };

    for (int c = tid; c < Cc; c += 256) {
        float xc  = xs[c];
        float xxc = xxs[c];
        #pragma unroll
        for (int f = 0; f < 5; ++f) {
            float acc = 0.f;
            #pragma unroll
            for (int d = 0; d < 32; ++d)
                acc += xxx[f * 32 + d] * maa_w2[(size_t)(f * 32 + d) * Cc + c];
            float o = xc + xxc * (maas[f][c] + acc);
            outs[f][(size_t)row * Cc + c] = __float2bfloat16(o);
        }
    }
}

// ============================================================
// Kernel 2: bf16 WMMA GEMM  C[MxN] = A[MxK](bf16) @ Bt[NxK](bf16)^T
// block tile 128x128, BK=64 (2 WMMA K-slices per stage),
// 8 waves (4M x 2N), wave tile 32x64.  epi: 0=none, 1=silu
// ============================================================
#define BM 128
#define BN 128
#define BK 64

__global__ __launch_bounds__(256)
void gemm_bf16_wmma_kernel(const __hip_bfloat16* __restrict__ A,
                           const __hip_bfloat16* __restrict__ Bt,  // [N][K]
                           float* __restrict__ Cout,
                           int M, int Nn, int K, int epi)
{
    __shared__ __align__(16) __hip_bfloat16 As[BM * BK];
    __shared__ __align__(16) __hip_bfloat16 Bs[BN * BK];

    const int tid   = threadIdx.x;
    const int lane  = tid & 31;
    const int wave  = tid >> 5;
    const int waveM = wave >> 1;     // 0..3
    const int waveN = wave & 1;      // 0..1
    const int m0 = blockIdx.y * BM;
    const int n0 = blockIdx.x * BN;

    v8f acc[2][4];
    #pragma unroll
    for (int i = 0; i < 2; ++i)
        #pragma unroll
        for (int j = 0; j < 4; ++j)
            #pragma unroll
            for (int e = 0; e < 8; ++e) acc[i][j][e] = 0.f;

    const int lm = lane & 15;
    const int kb = (lane < 16) ? 0 : 8;   // per-lane K base (16-bit layout)

    // staging coordinates (8 x b128 per row, 32 rows per pass)
    const int sr = tid >> 3;              // 0..31 row group
    const int sc = (tid & 7) * 8;         // 0..56

    for (int k0 = 0; k0 < K; k0 += BK) {
        // ---- stage A and B tiles: pure b128 copies (or async-to-LDS) ----
        #pragma unroll
        for (int it = 0; it < 4; ++it) {
            int r = sr + it * 32;
            const __hip_bfloat16* ga = A  + (size_t)(m0 + r) * K + k0 + sc;
            const __hip_bfloat16* gb = Bt + (size_t)(n0 + r) * K + k0 + sc;
#if HAS_ASYNC_LDS
            __builtin_amdgcn_global_load_async_to_lds_b128(
                (__attribute__((address_space(1))) v4i_vs*)ga,
                (__attribute__((address_space(3))) v4i_vs*)&As[r * BK + sc], 0, 0);
            __builtin_amdgcn_global_load_async_to_lds_b128(
                (__attribute__((address_space(1))) v4i_vs*)gb,
                (__attribute__((address_space(3))) v4i_vs*)&Bs[r * BK + sc], 0, 0);
#else
            *(uint4*)&As[r * BK + sc] = *(const uint4*)ga;
            *(uint4*)&Bs[r * BK + sc] = *(const uint4*)gb;
#endif
        }
        if (k0 + BK < K) {  // prefetch next K-tile into cache
            __builtin_prefetch(A  + (size_t)(m0 + sr) * K + k0 + BK + sc, 0, 1);
            __builtin_prefetch(Bt + (size_t)(n0 + sr) * K + k0 + BK + sc, 0, 1);
        }
#if HAS_ASYNC_LDS
        wait_asynccnt0();
#endif
        __syncthreads();

        // ---- two K-slices of 32 per stage ----
        #pragma unroll
        for (int ks = 0; ks < 2; ++ks) {
            const int kofs = ks * 32 + kb;
            v16bf afrag[2], bfrag[4];
            #pragma unroll
            for (int mt = 0; mt < 2; ++mt) {
                int mr = waveM * 32 + mt * 16 + lm;
                FragU f;
                f.u[0] = *(const uint4*)(&As[mr * BK + kofs]);
                f.u[1] = *(const uint4*)(&As[mr * BK + 16 + kofs]);
                afrag[mt] = f.v;
            }
            #pragma unroll
            for (int nt = 0; nt < 4; ++nt) {
                int nc = waveN * 64 + nt * 16 + lm;
                FragU f;
                f.u[0] = *(const uint4*)(&Bs[nc * BK + kofs]);
                f.u[1] = *(const uint4*)(&Bs[nc * BK + 16 + kofs]);
                bfrag[nt] = f.v;
            }
            #pragma unroll
            for (int mt = 0; mt < 2; ++mt)
                #pragma unroll
                for (int nt = 0; nt < 4; ++nt)
                    acc[mt][nt] = __builtin_amdgcn_wmma_f32_16x16x32_bf16(
                        false, afrag[mt], false, bfrag[nt],
                        (short)0, acc[mt][nt], false, false);
        }
        __syncthreads();
    }

    // ---- epilogue: lane l, vgpr g -> row = g + 8*(l>>4), col = l&15 ----
    const int rofs = (lane >> 4) * 8;
    #pragma unroll
    for (int mt = 0; mt < 2; ++mt) {
        #pragma unroll
        for (int nt = 0; nt < 4; ++nt) {
            int col = n0 + waveN * 64 + nt * 16 + lm;
            #pragma unroll
            for (int g = 0; g < 8; ++g) {
                int r = m0 + waveM * 32 + mt * 16 + rofs + g;
                float v = acc[mt][nt][g];
                if (epi == 1)
                    v = v * __builtin_amdgcn_rcpf(1.f + __expf(-v));  // fast silu
                Cout[(size_t)r * Nn + col] = v;
            }
        }
    }
}

// ============================================================
// Kernel 3: decay = exp(-exp(time_decay + tanh(xw@w1)@w2))
// ============================================================
__global__ __launch_bounds__(64)
void decay_kernel(const __hip_bfloat16* __restrict__ xw,
                  const float* __restrict__ time_decay,  // (C)
                  const float* __restrict__ w1,          // (C,64)
                  const float* __restrict__ w2,          // (64,C)
                  float* __restrict__ decay)
{
    __shared__ float xrow[Cc];
    __shared__ float hidden[64];
    const int row = blockIdx.x;
    const int tid = threadIdx.x;

    for (int c = tid; c < Cc; c += 64)
        xrow[c] = __bfloat162float(xw[(size_t)row * Cc + c]);
    __syncthreads();

    float s = 0.f;
    for (int c = 0; c < Cc; ++c)
        s += xrow[c] * w1[(size_t)c * 64 + tid];
    hidden[tid] = tanhf(s);
    __syncthreads();

    for (int c = tid; c < Cc; c += 64) {
        float a = time_decay[c];
        #pragma unroll
        for (int d = 0; d < 64; ++d)
            a += hidden[d] * w2[(size_t)d * Cc + c];
        decay[(size_t)row * Cc + c] = __expf(-__expf(a));
    }
}

// ============================================================
// Kernel 4: WKV scan. one block per (b,h), 64 threads, thread = column v.
// S register-resident; double-buffered LDS broadcast -> 1 barrier/step.
// ============================================================
__global__ __launch_bounds__(64)
void wkv_scan_kernel(const float* __restrict__ rb,
                     const float* __restrict__ kb_,
                     const float* __restrict__ vb,
                     const float* __restrict__ db,
                     const float* __restrict__ wkv_state,  // (B,H,N,N)
                     const float* __restrict__ u,          // (H,N)
                     float* __restrict__ out,              // (B,T,C)
                     float* __restrict__ Sfinal)           // (B,H,N,N)
{
    __shared__ float rr[2][Nh], kk[2][Nh], dd[2][Nh], us[Nh];
    const int b   = blockIdx.x / Hc;
    const int h   = blockIdx.x % Hc;
    const int tid = threadIdx.x;      // v index

    float S[Nh];
    #pragma unroll
    for (int k2 = 0; k2 < Nh; ++k2)
        S[k2] = wkv_state[((size_t)(b * Hc + h) * Nh + k2) * Nh + tid];

    us[tid] = u[h * Nh + tid];

    for (int t = 0; t < Tc; ++t) {
        const int    pb   = t & 1;
        const size_t base = (size_t)(b * Tc + t) * Cc + h * Nh;
        rr[pb][tid] = rb [base + tid];
        kk[pb][tid] = kb_[base + tid];
        dd[pb][tid] = db [base + tid];
        __syncthreads();

        const float vv = vb[base + tid];
        float o = 0.f;
        #pragma unroll
        for (int k2 = 0; k2 < Nh; ++k2) {
            float kvk = kk[pb][k2] * vv;
            o += rr[pb][k2] * (S[k2] + us[k2] * kvk);
            S[k2] = dd[pb][k2] * S[k2] + kvk;
        }
        out[base + tid] = o;
        // no trailing barrier: next step writes the other buffer
    }

    #pragma unroll
    for (int k2 = 0; k2 < Nh; ++k2)
        Sfinal[((size_t)(b * Hc + h) * Nh + k2) * Nh + tid] = S[k2];
}

// ============================================================
// Kernel 5: per-head groupnorm + affine + gate -> bf16 y
// ============================================================
__global__ __launch_bounds__(64)
void gn_gate_kernel(const float* __restrict__ wkv,
                    const float* __restrict__ gbuf,
                    const float* __restrict__ ln_g,
                    const float* __restrict__ ln_b,
                    __hip_bfloat16* __restrict__ y)
{
    __shared__ float red[Nh];
    const int grp = blockIdx.x;       // (b*T+t)*H + h
    const int bt  = grp / Hc;
    const int h   = grp % Hc;
    const int tid = threadIdx.x;
    const int c   = h * Nh + tid;
    const size_t idx = (size_t)bt * Cc + c;

    const float val = wkv[idx];

    red[tid] = val; __syncthreads();
    #pragma unroll
    for (int s2 = 32; s2 > 0; s2 >>= 1) {
        if (tid < s2) red[tid] += red[tid + s2];
        __syncthreads();
    }
    const float mean = red[0] * (1.f / Nh);
    __syncthreads();

    const float dv = val - mean;
    red[tid] = dv * dv; __syncthreads();
    #pragma unroll
    for (int s2 = 32; s2 > 0; s2 >>= 1) {
        if (tid < s2) red[tid] += red[tid + s2];
        __syncthreads();
    }
    const float var = red[0] * (1.f / Nh);

    const float eps = 1e-5f * 64.f;
    float o = dv * rsqrtf(var + eps) * ln_g[c] + ln_b[c];
    o *= gbuf[idx];
    y[idx] = __float2bfloat16(o);
}

// ============================================================
// Kernel 6: new_shift = x[:, T-1, :]
// ============================================================
__global__ __launch_bounds__(256)
void shift_copy_kernel(const float* __restrict__ x, float* __restrict__ dst)
{
    int i = blockIdx.x * 256 + threadIdx.x;
    if (i < Bc * Cc) {
        int b = i / Cc, c = i % Cc;
        dst[i] = x[((size_t)b * Tc + (Tc - 1)) * Cc + c];
    }
}

// ============================================================
// Launcher
// ============================================================
extern "C" void kernel_launch(void* const* d_in, const int* in_sizes, int n_in,
                              void* d_out, int out_size, void* d_ws, size_t ws_size,
                              hipStream_t stream)
{
    const float* x           = (const float*)d_in[0];
    const float* shift_state = (const float*)d_in[1];
    const float* wkv_state   = (const float*)d_in[2];
    const float* maa_x       = (const float*)d_in[3];
    const float* maa_w       = (const float*)d_in[4];
    const float* maa_k       = (const float*)d_in[5];
    const float* maa_v       = (const float*)d_in[6];
    const float* maa_r       = (const float*)d_in[7];
    const float* maa_g       = (const float*)d_in[8];
    const float* maa_w1      = (const float*)d_in[9];
    const float* maa_w2      = (const float*)d_in[10];
    const float* time_decay  = (const float*)d_in[11];
    const float* decay_w1    = (const float*)d_in[12];
    const float* decay_w2    = (const float*)d_in[13];
    const float* time_faaaa  = (const float*)d_in[14];
    const float* Wr          = (const float*)d_in[15];
    const float* Wk          = (const float*)d_in[16];
    const float* Wv          = (const float*)d_in[17];
    const float* Wg          = (const float*)d_in[18];
    const float* Wo          = (const float*)d_in[19];
    const float* ln_g        = (const float*)d_in[20];
    const float* ln_b        = (const float*)d_in[21];

    float* out_main  = (float*)d_out;                       // (B,T,C)
    float* out_shift = out_main + OUT_ELEMS;                // (B,1,C)
    float* out_S     = out_shift + SHIFT_ELEMS;             // (B,H,N,N)

    // ---- workspace carve-up ----
    char* ws = (char*)d_ws;
    size_t off = 0;
    auto carve = [&](size_t bytes) -> void* {
        off = (off + 255) & ~(size_t)255;
        void* p = ws + off;
        off += bytes;
        return p;
    };
    const size_t MC = (size_t)Mrows * Cc;
    __hip_bfloat16* Wt_all = (__hip_bfloat16*)carve(5 * MC * 2);  // Wr,Wk,Wv,Wg,Wo ^T bf16
    __hip_bfloat16* xw = (__hip_bfloat16*)carve(MC * 2);
    __hip_bfloat16* xk = (__hip_bfloat16*)carve(MC * 2);
    __hip_bfloat16* xv = (__hip_bfloat16*)carve(MC * 2);
    __hip_bfloat16* xr = (__hip_bfloat16*)carve(MC * 2);
    __hip_bfloat16* xg = (__hip_bfloat16*)carve(MC * 2);
    float* rbuf   = (float*)carve(MC * 4);
    float* kbuf   = (float*)carve(MC * 4);
    float* vbuf   = (float*)carve(MC * 4);
    float* gbuf   = (float*)carve(MC * 4);
    float* dbuf   = (float*)carve(MC * 4);
    float* wkvbuf = (float*)carve(MC * 4);
    __hip_bfloat16* ybuf = (__hip_bfloat16*)carve(MC * 2);
    (void)ws_size; (void)in_sizes; (void)n_in; (void)out_size;

    // 0. transpose+convert all five weights
    wtrans_kernel<<<dim3(Cc / 32, Cc / 32, 5), 256, 0, stream>>>(
        Wr, Wk, Wv, Wg, Wo, Wt_all);

    // 1. mixing
    rwkv_mix_kernel<<<Bc * Tc, 256, 0, stream>>>(
        x, shift_state, maa_x, maa_w, maa_k, maa_v, maa_r, maa_g,
        maa_w1, maa_w2, xw, xk, xv, xr, xg);

    // 2. four big WMMA GEMMs (A bf16 @ Wt^T bf16 -> f32)
    dim3 ggrid(Cc / BN, Mrows / BM);
    gemm_bf16_wmma_kernel<<<ggrid, 256, 0, stream>>>(xr, Wt_all + 0 * MC, rbuf, Mrows, Cc, Cc, 0);
    gemm_bf16_wmma_kernel<<<ggrid, 256, 0, stream>>>(xk, Wt_all + 1 * MC, kbuf, Mrows, Cc, Cc, 0);
    gemm_bf16_wmma_kernel<<<ggrid, 256, 0, stream>>>(xv, Wt_all + 2 * MC, vbuf, Mrows, Cc, Cc, 0);
    gemm_bf16_wmma_kernel<<<ggrid, 256, 0, stream>>>(xg, Wt_all + 3 * MC, gbuf, Mrows, Cc, Cc, 1);

    // 3. decay path
    decay_kernel<<<Bc * Tc, 64, 0, stream>>>(xw, time_decay, decay_w1, decay_w2, dbuf);

    // 4. WKV sequential scan (writes S_final into d_out)
    wkv_scan_kernel<<<Bc * Hc, 64, 0, stream>>>(
        rbuf, kbuf, vbuf, dbuf, wkv_state, time_faaaa, wkvbuf, out_S);

    // 5. groupnorm + gate
    gn_gate_kernel<<<Bc * Tc * Hc, 64, 0, stream>>>(wkvbuf, gbuf, ln_g, ln_b, ybuf);

    // 6. output projection (WMMA) straight into d_out
    gemm_bf16_wmma_kernel<<<ggrid, 256, 0, stream>>>(ybuf, Wt_all + 4 * MC, out_main, Mrows, Cc, Cc, 0);

    // 7. new shift state
    shift_copy_kernel<<<(Bc * Cc + 255) / 256, 256, 0, stream>>>(x, out_shift);
}
